// MemoryEfficientVocabOutput_12859132084435
// MI455X (gfx1250) — matrix operand order repaired and checked by reading, not compiled, not run on hardware
//
#include <hip/hip_runtime.h>
#include <hip/hip_bf16.h>
#include <math.h>

#define TOKENS 4096
#define DMODEL 1024
#define VOCAB  50257
#define NCHUNK 16
#define CHUNK  3200        // 25 * 128; NCHUNK*CHUNK >= VOCAB
#define MTILE  128         // tokens per block (8 waves x 16 rows)
#define NSTEP  128         // vocab rows per LDS tile
#define KSTEP  32
#define LDS_STRIDE 40      // bf16 elems per LDS row: 80B pitch (16B aligned, 20-bank stride)

typedef __attribute__((ext_vector_type(16))) __bf16 v16bf;
typedef __attribute__((ext_vector_type(8)))  float  v8f;

union Frag16 { v16bf v; uint4 q[2]; };

__device__ __forceinline__ unsigned short f32_to_bf16_rne(float f) {
  unsigned int u = __builtin_bit_cast(unsigned int, f);
  u += 0x7FFFu + ((u >> 16) & 1u);
  return (unsigned short)(u >> 16);
}
__device__ __forceinline__ unsigned int pack2(float lo, float hi) {
  return (unsigned int)f32_to_bf16_rne(lo) | ((unsigned int)f32_to_bf16_rne(hi) << 16);
}

// ---- Kernel 1: fp32 input -> bf16 (row-major [TOKENS][DMODEL]) --------------
__global__ void cvt_input_bf16(const float* __restrict__ in,
                               unsigned short* __restrict__ out) {
  const int i = (blockIdx.x * blockDim.x + threadIdx.x) * 8;
  const float4 f0 = *(const float4*)(in + i);
  const float4 f1 = *(const float4*)(in + i + 4);
  uint4 o;
  o.x = pack2(f0.x, f0.y); o.y = pack2(f0.z, f0.w);
  o.z = pack2(f1.x, f1.y); o.w = pack2(f1.z, f1.w);
  *(uint4*)(out + i) = o;
}

// ---- Kernel 2: tiled WMMA GEMM + online logsumexp partials ------------------
__global__ __launch_bounds__(256)
void vocab_lse_partial(const unsigned short* __restrict__ inB,
                       const float* __restrict__ W,
                       float* __restrict__ part) {
  __shared__ __align__(16) unsigned short bt[NSTEP * LDS_STRIDE];

  const int tid  = threadIdx.x;
  const int lane = tid & 31;
  const int wave = tid >> 5;
  const int l16  = lane & 15;
  const int kh   = lane >> 4;          // K-half per WMMA 16-bit layout

  const int tokBase = blockIdx.x * MTILE;
  const int chunk   = blockIdx.y;
  const int cbeg    = chunk * CHUNK;
  const int cend    = min(cbeg + CHUNK, VOCAB);

  const int myTok = tokBase + wave * 16 + l16;      // A-frag row for this lane
  const unsigned short* aRow = inB + (size_t)myTok * DMODEL;

  float mrun[8], srun[8];
#pragma unroll
  for (int v = 0; v < 8; ++v) { mrun[v] = -INFINITY; srun[v] = 0.f; }

  // cooperative LDS-fill coordinates: 2 threads per vocab row, 16 floats each
  const int frow = tid >> 1;
  const int fcol = (tid & 1) * 16;

  for (int vs = cbeg; vs < cend; vs += NSTEP) {
    v8f acc[8];
    const v8f zero = {0.f,0.f,0.f,0.f,0.f,0.f,0.f,0.f};
#pragma unroll
    for (int nt = 0; nt < 8; ++nt) acc[nt] = zero;

    for (int k = 0; k < DMODEL; k += KSTEP) {
      // -- fp32 weights -> bf16 LDS tile [128 x 32] --
      {
        const int n = vs + frow;
        uint4 o0, o1;
        if (n < VOCAB) {
          const float* wp = W + (size_t)n * DMODEL + k + fcol;
          const float4 f0 = *(const float4*)(wp + 0);
          const float4 f1 = *(const float4*)(wp + 4);
          const float4 f2 = *(const float4*)(wp + 8);
          const float4 f3 = *(const float4*)(wp + 12);
          o0.x = pack2(f0.x, f0.y); o0.y = pack2(f0.z, f0.w);
          o0.z = pack2(f1.x, f1.y); o0.w = pack2(f1.z, f1.w);
          o1.x = pack2(f2.x, f2.y); o1.y = pack2(f2.z, f2.w);
          o1.z = pack2(f3.x, f3.y); o1.w = pack2(f3.z, f3.w);
          if (k + KSTEP < DMODEL)
            __builtin_prefetch(wp + KSTEP, 0, 3);   // global_prefetch_b8
        } else {
          o0 = make_uint4(0u, 0u, 0u, 0u); o1 = o0;
        }
        *(uint4*)&bt[frow * LDS_STRIDE + fcol]     = o0;
        *(uint4*)&bt[frow * LDS_STRIDE + fcol + 8] = o1;
      }
      __syncthreads();

      // -- A fragment: 16 tokens x 32 K (bf16), per 7.12.2 16-bit A layout --
      Frag16 a;
      a.q[0] = *(const uint4*)(aRow + k + kh * 8);
      a.q[1] = *(const uint4*)(aRow + k + 16 + kh * 8);

      // -- 8 WMMAs covering the 128-wide vocab tile --
#pragma unroll
      for (int nt = 0; nt < 8; ++nt) {
        Frag16 b;
        const unsigned short* bp = &bt[(nt * 16 + l16) * LDS_STRIDE + kh * 8];
        b.q[0] = *(const uint4*)(bp);
        b.q[1] = *(const uint4*)(bp + 16);
        acc[nt] = __builtin_amdgcn_wmma_f32_16x16x32_bf16(
            false, a.v, false, b.v, (short)0, acc[nt], false, false);
      }
      __syncthreads();
    }

    // -- online logsumexp: fold this 128-vocab tile into per-lane run state --
    const int colbase = vs + l16;
#pragma unroll
    for (int v = 0; v < 8; ++v) {
      float mloc = -INFINITY;
#pragma unroll
      for (int nt = 0; nt < 8; ++nt)
        if (colbase + nt * 16 < VOCAB) mloc = fmaxf(mloc, acc[nt][v]);
      if (mloc != -INFINITY) {
        const float mn = fmaxf(mrun[v], mloc);
        float s = srun[v] * __expf(mrun[v] - mn);
#pragma unroll
        for (int nt = 0; nt < 8; ++nt)
          if (colbase + nt * 16 < VOCAB) s += __expf(acc[nt][v] - mn);
        mrun[v] = mn; srun[v] = s;
      }
    }
  }

  // -- combine the 16 lanes that hold the same C row (logsumexp merge) --
#pragma unroll
  for (int v = 0; v < 8; ++v) {
    float m = mrun[v], s = srun[v];
#pragma unroll
    for (int off = 8; off >= 1; off >>= 1) {
      const float mo = __shfl_xor(m, off, 32);
      const float so = __shfl_xor(s, off, 32);
      const float mn = fmaxf(m, mo);
      s = s * __expf(m - mn) + so * __expf(mo - mn);
      m = mn;
    }
    if (l16 == 0) {
      const int tok = tokBase + wave * 16 + v + kh * 8;
      part[(size_t)(chunk * TOKENS + tok) * 2 + 0] = m;
      part[(size_t)(chunk * TOKENS + tok) * 2 + 1] = s;
    }
  }
}

// ---- Kernel 3: per-token target logit (gathered fp32 dot) -------------------
__global__ void target_score(const float* __restrict__ inp,
                             const float* __restrict__ W,
                             const int* __restrict__ tgt,
                             float* __restrict__ ts) {
  const int lane = threadIdx.x & 31;
  const int wave = threadIdx.x >> 5;
  const int t = blockIdx.x * 8 + wave;
  const float* a = inp + (size_t)t * DMODEL;
  const float* b = W + (size_t)tgt[t] * DMODEL;
  float sum = 0.f;
  for (int i = lane * 4; i < DMODEL; i += 128) {
    const float4 av = *(const float4*)(a + i);
    const float4 bv = *(const float4*)(b + i);
    sum += av.x * bv.x + av.y * bv.y + av.z * bv.z + av.w * bv.w;
  }
#pragma unroll
  for (int off = 16; off >= 1; off >>= 1) sum += __shfl_xor(sum, off, 32);
  if (lane == 0) ts[t] = sum;
}

// ---- Kernel 4: merge chunk partials, sum loss -------------------------------
__global__ void finalize_loss(const float* __restrict__ part,
                              const float* __restrict__ ts,
                              float* __restrict__ out) {
  __shared__ float red[256];
  const int tid = threadIdx.x;
  float local = 0.f;
  for (int t = tid; t < TOKENS; t += 256) {
    float m = -INFINITY, s = 0.f;
    for (int c = 0; c < NCHUNK; ++c) {
      const float pm = part[(size_t)(c * TOKENS + t) * 2 + 0];
      const float ps = part[(size_t)(c * TOKENS + t) * 2 + 1];
      const float mn = fmaxf(m, pm);
      s = s * __expf(m - mn) + ps * __expf(pm - mn);
      m = mn;
    }
    local += (m + __logf(s)) - ts[t];
  }
  red[tid] = local;
  __syncthreads();
  for (int w = 128; w > 0; w >>= 1) {
    if (tid < w) red[tid] += red[tid + w];
    __syncthreads();
  }
  if (tid == 0) out[0] = red[0];
}

extern "C" void kernel_launch(void* const* d_in, const int* in_sizes, int n_in,
                              void* d_out, int out_size, void* d_ws, size_t ws_size,
                              hipStream_t stream) {
  (void)in_sizes; (void)n_in; (void)out_size; (void)ws_size;
  const float* inp = (const float*)d_in[0];
  const float* W   = (const float*)d_in[1];
  const int*   tgt = (const int*)d_in[2];
  float* out = (float*)d_out;

  unsigned short* inB = (unsigned short*)d_ws;
  const size_t IN_BYTES   = (size_t)TOKENS * DMODEL * 2;            // 8 MiB
  float* part = (float*)((char*)d_ws + IN_BYTES);
  const size_t PART_BYTES = (size_t)NCHUNK * TOKENS * 2 * sizeof(float);
  float* ts   = (float*)((char*)d_ws + IN_BYTES + PART_BYTES);

  cvt_input_bf16<<<(TOKENS * DMODEL / 8) / 256, 256, 0, stream>>>(inp, inB);

  dim3 grid(TOKENS / MTILE, NCHUNK);
  vocab_lse_partial<<<grid, 256, 0, stream>>>(inB, W, part);

  target_score<<<TOKENS / 8, 256, 0, stream>>>(inp, W, tgt, ts);

  finalize_loss<<<1, 256, 0, stream>>>(part, ts, out);
}
